// FlowFaceCrossAttentionBlock_34282428956919
// MI455X (gfx1250) — matrix operand order repaired and checked by reading, not compiled, not run on hardware
//
#include <hip/hip_runtime.h>

typedef __attribute__((ext_vector_type(16))) __bf16 v16bf;
typedef __attribute__((ext_vector_type(8)))  float  v8f;

union Frag { v16bf v; unsigned u[8]; };

static constexpr int SEQ  = 1024;   // S
static constexpr int DIM  = 1024;   // QD = KVD = H*128
static constexpr int BATCH = 8;
static constexpr int HEADS = 8;
static constexpr int HD   = 128;    // head dim
static constexpr int MROWS = BATCH * SEQ; // 8192

__device__ __forceinline__ unsigned short f2bf(float f) {
  unsigned u = __float_as_uint(f);
  unsigned r = u + 0x7FFFu + ((u >> 16) & 1u);
  return (unsigned short)(r >> 16);
}
__device__ __forceinline__ float bf2f(unsigned short b) {
  return __uint_as_float(((unsigned)b) << 16);
}
__device__ __forceinline__ v8f vzero8() {
  v8f r;
  #pragma unroll
  for (int i = 0; i < 8; ++i) r[i] = 0.0f;
  return r;
}

// ---- CDNA5 async global->LDS copies (ASYNCcnt tracked) ----
__device__ __forceinline__ void async_ld_b64(unsigned lds_off, const void* g) {
  unsigned long long ga = (unsigned long long)g;
  asm volatile("global_load_async_to_lds_b64 %0, %1, off" :: "v"(lds_off), "v"(ga) : "memory");
}
__device__ __forceinline__ void async_ld_b128(unsigned lds_off, const void* g) {
  unsigned long long ga = (unsigned long long)g;
  asm volatile("global_load_async_to_lds_b128 %0, %1, off" :: "v"(lds_off), "v"(ga) : "memory");
}
__device__ __forceinline__ void wait_async0() {
  asm volatile("s_wait_asynccnt 0x0" ::: "memory");
}

// ---------------- fp32 -> bf16 convert ----------------
__global__ void __launch_bounds__(256) k_f2b(const float* __restrict__ src,
                                             unsigned short* __restrict__ dst, int n) {
  int i = blockIdx.x * 256 + threadIdx.x;
  if (i < n) dst[i] = f2bf(src[i]);
}

// ---------------- GEMM: C = act(A @ W^T + bias) [+ Res] ----------------
// A: [M,K] bf16 row-major. W: [N,K] bf16 row-major (torch Linear layout).
// 4 waves, block tile 128(M) x 128(N), K-step 32, double-buffered async LDS tiles.
// Each wave: 32 rows x 128 cols -> 16 WMMA per K-step.
#define LDS_STRIDE 36
__device__ __forceinline__ void gemm_issue_tile(const unsigned short* A, const unsigned short* W,
                                                int m0, int n0, int K, int k0,
                                                unsigned ldsA, unsigned ldsB, int tid) {
  const int r0 = tid >> 3;   // 0..15
  const int p  = tid & 7;    // 0..7 (8-byte chunk within 32-elem row)
  #pragma unroll
  for (int i = 0; i < 8; ++i)
    async_ld_b64(ldsA + (unsigned)(((r0 + 16 * i) * LDS_STRIDE + p * 4) * 2),
                 A + (size_t)(m0 + r0 + 16 * i) * K + k0 + p * 4);
  #pragma unroll
  for (int i = 0; i < 8; ++i)
    async_ld_b64(ldsB + (unsigned)(((r0 + 16 * i) * LDS_STRIDE + p * 4) * 2),
                 W + (size_t)(n0 + r0 + 16 * i) * K + k0 + p * 4);
}

__device__ __forceinline__ void load_bfrag(Frag& f, const unsigned short* pB, int t, int h, int ln) {
  int col = t * 16 + ln;
  #pragma unroll
  for (int d = 0; d < 8; ++d) {
    int kk = ((d >> 2) * 16) + h * 8 + 2 * (d & 3); // 16-bit B layout
    f.u[d] = *(const unsigned*)(pB + col * LDS_STRIDE + kk);
  }
}

__global__ void __launch_bounds__(128, 1)
k_gemm(const unsigned short* __restrict__ A,
       const unsigned short* __restrict__ W,
       const float* __restrict__ bias,
       const unsigned short* __restrict__ Res,
       unsigned short* __restrict__ Cb,
       float* __restrict__ Cf,
       int M, int N, int K, int act) {
  __shared__ unsigned short sA[2][128 * LDS_STRIDE];
  __shared__ unsigned short sB[2][128 * LDS_STRIDE];

  const int tid  = threadIdx.x;        // 0..127
  const int wave = tid >> 5;
  const int lane = tid & 31;
  const int h    = lane >> 4;
  const int ln   = lane & 15;
  const int m0   = blockIdx.y * 128;
  const int n0   = blockIdx.x * 128;

  const unsigned baseA[2] = {(unsigned)(unsigned long long)&sA[0][0],
                             (unsigned)(unsigned long long)&sA[1][0]};
  const unsigned baseB[2] = {(unsigned)(unsigned long long)&sB[0][0],
                             (unsigned)(unsigned long long)&sB[1][0]};

  v8f acc[2][8];
  #pragma unroll
  for (int s = 0; s < 2; ++s)
    #pragma unroll
    for (int t = 0; t < 8; ++t) acc[s][t] = vzero8();

  // prologue: async-copy first K tile
  gemm_issue_tile(A, W, m0, n0, K, 0, baseA[0], baseB[0], tid);
  wait_async0();
  __syncthreads();

  for (int k0 = 0; k0 < K; k0 += 32) {
    const int buf = (k0 >> 5) & 1;
    if (k0 + 32 < K)   // overlap: async-copy next tile while computing this one
      gemm_issue_tile(A, W, m0, n0, K, k0 + 32, baseA[buf ^ 1], baseB[buf ^ 1], tid);

    const unsigned short* pA = sA[buf];
    const unsigned short* pB = sB[buf];

    Frag af[2];
    #pragma unroll
    for (int s = 0; s < 2; ++s) {
      int row = wave * 32 + s * 16 + ln;
      #pragma unroll
      for (int d = 0; d < 8; ++d) {
        int kk = ((d >> 2) * 16) + h * 8 + 2 * (d & 3); // 16-bit A layout
        af[s].u[d] = *(const unsigned*)(pA + row * LDS_STRIDE + kk);
      }
    }
    // B-fragment double rotation to break WMMA->VALU WAR hazards
    Frag bfr[2];
    load_bfrag(bfr[0], pB, 0, h, ln);
    #pragma unroll
    for (int t = 0; t < 8; ++t) {
      if (t < 7) load_bfrag(bfr[(t + 1) & 1], pB, t + 1, h, ln);
      const Frag& cb = bfr[t & 1];
      acc[0][t] = __builtin_amdgcn_wmma_f32_16x16x32_bf16(
          false, af[0].v, false, cb.v, (short)0, acc[0][t], false, false);
      acc[1][t] = __builtin_amdgcn_wmma_f32_16x16x32_bf16(
          false, af[1].v, false, cb.v, (short)0, acc[1][t], false, false);
    }
    wait_async0();
    __syncthreads();
  }

  // epilogue: bias + optional leaky_relu + optional residual
  #pragma unroll
  for (int s = 0; s < 2; ++s)
    #pragma unroll
    for (int t = 0; t < 8; ++t)
      #pragma unroll
      for (int v = 0; v < 8; ++v) {
        int m = m0 + wave * 32 + s * 16 + v + 8 * h;
        int n = n0 + t * 16 + ln;
        float val = acc[s][t][v] + bias[n];
        if (act) val = (val > 0.0f) ? val : val * 0.01f;
        size_t idx = (size_t)m * N + n;
        if (Res) val += bf2f(Res[idx]);
        if (Cb) Cb[idx] = f2bf(val);
        if (Cf) Cf[idx] = val;
      }
}

// ---------------- Fused attention: softmax(Q K^T * scale) @ V (+QVadd) ----------------
// One wave per (b, head, 16-query-row tile). Q/K/V in merged [B,S,1024] bf16.
__global__ void __launch_bounds__(32, 1)
k_attn(const unsigned short* __restrict__ Q,
       const unsigned short* __restrict__ Kb,
       const unsigned short* __restrict__ Vb,
       const unsigned short* __restrict__ QVadd,
       unsigned short* __restrict__ O,
       float scale) {
  extern __shared__ char smem[];
  float*          sS    = (float*)smem;                            // 16x1024 f32 (64KB)
  unsigned short* sP    = (unsigned short*)(smem + 65536);         // 16x1024 bf16 (32KB)
  unsigned short* sV    = (unsigned short*)(smem + 65536 + 32768); // 2 x (32x128) bf16 (16KB)
  float*          sStat = (float*)(smem + 65536 + 32768 + 16384);  // 64 floats

  const int lane = threadIdx.x;
  const int h  = lane >> 4;
  const int ln = lane & 15;
  const int bidx = blockIdx.x;
  const int qt   = bidx & 63;
  const int head = (bidx >> 6) & 7;
  const int b    = bidx >> 9;
  const size_t rowbase = (size_t)b * SEQ * DIM;
  const int hoff = head * HD;

  // Q fragments: 4 K-chunks of 32 along head dim
  Frag qf[4];
  {
    const unsigned short* qp = Q + rowbase + (size_t)(qt * 16 + ln) * DIM + hoff;
    #pragma unroll
    for (int c = 0; c < 4; ++c)
      #pragma unroll
      for (int d = 0; d < 8; ++d) {
        int kk = c * 32 + ((d >> 2) * 16) + h * 8 + 2 * (d & 3);
        qf[c].u[d] = *(const unsigned*)(qp + kk);
      }
  }

  // Phase 1: full 16x1024 score strip via WMMA
  for (int j0 = 0; j0 < SEQ; j0 += 16) {
    v8f acc = vzero8();
    const unsigned short* kp = Kb + rowbase + (size_t)(j0 + ln) * DIM + hoff;
    #pragma unroll
    for (int c = 0; c < 4; ++c) {
      Frag bfr;
      #pragma unroll
      for (int d = 0; d < 8; ++d)
        bfr.u[d] = *(const unsigned*)(kp + c * 32 + ((d >> 2) * 16) + h * 8 + 2 * (d & 3));
      acc = __builtin_amdgcn_wmma_f32_16x16x32_bf16(
          false, qf[c].v, false, bfr.v, (short)0, acc, false, false);
    }
    #pragma unroll
    for (int v = 0; v < 8; ++v)
      sS[(v + 8 * h) * SEQ + j0 + ln] = acc[v] * scale;
  }
  __syncthreads();

  // Phase 2: softmax. Row = ln, column half = h (512 cols each).
  {
    const int roff = ln * SEQ + h * 512;
    float m = -3.0e38f;
    for (int i = 0; i < 512; ++i) m = fmaxf(m, sS[roff + i]);
    sStat[lane] = m;
    __syncthreads();
    float rm = fmaxf(sStat[ln], sStat[16 + ln]);
    __syncthreads();
    float sum = 0.0f;
    for (int i = 0; i < 512; ++i) {
      float e = __expf(sS[roff + i] - rm);
      sP[roff + i] = f2bf(e);
      sum += e;
    }
    sStat[lane] = sum;
    __syncthreads();
    float rinv = 1.0f / (sStat[ln] + sStat[16 + ln]);
    if (h == 0) sStat[32 + ln] = rinv;
    __syncthreads();
  }

  // Phase 3: O = P @ V with double-buffered async V tiles (32 keys x 128 dims each)
  const unsigned baseV[2] = {(unsigned)(unsigned long long)sV,
                             (unsigned)(unsigned long long)(sV + 32 * HD)};
  v8f ao[8];
  #pragma unroll
  for (int t = 0; t < 8; ++t) ao[t] = vzero8();

  // prologue: async-copy first V tile
  {
    const unsigned short* vp = Vb + rowbase + (size_t)lane * DIM + hoff;
    #pragma unroll
    for (int p = 0; p < 16; ++p)
      async_ld_b128(baseV[0] + (unsigned)((lane * HD + p * 8) * 2), vp + p * 8);
    wait_async0();
    __syncthreads();
  }

  for (int j0 = 0; j0 < SEQ; j0 += 32) {
    const int buf = (j0 >> 5) & 1;
    if (j0 + 32 < SEQ) {   // overlap next V tile copy with WMMA compute
      const unsigned short* vp = Vb + rowbase + (size_t)(j0 + 32 + lane) * DIM + hoff;
      #pragma unroll
      for (int p = 0; p < 16; ++p)
        async_ld_b128(baseV[buf ^ 1] + (unsigned)((lane * HD + p * 8) * 2), vp + p * 8);
    }
    const unsigned short* pV = sV + buf * 32 * HD;

    Frag pf;
    #pragma unroll
    for (int d = 0; d < 8; ++d) {
      int kk = ((d >> 2) * 16) + h * 8 + 2 * (d & 3);
      pf.u[d] = *(const unsigned*)(sP + ln * SEQ + j0 + kk);
    }
    #pragma unroll
    for (int t = 0; t < 8; ++t) {
      Frag vf;
      int n = t * 16 + ln;
      #pragma unroll
      for (int d = 0; d < 8; ++d) {
        int k2 = ((d >> 2) * 16) + h * 8 + 2 * (d & 3);
        unsigned lo = pV[k2 * HD + n];
        unsigned hi = pV[(k2 + 1) * HD + n];
        vf.u[d] = lo | (hi << 16);
      }
      ao[t] = __builtin_amdgcn_wmma_f32_16x16x32_bf16(
          false, pf.v, false, vf.v, (short)0, ao[t], false, false);
    }
    wait_async0();
    __syncthreads();
  }

  // Epilogue: scale by 1/rowsum, optional +qv, store bf16
  #pragma unroll
  for (int t = 0; t < 8; ++t)
    #pragma unroll
    for (int v = 0; v < 8; ++v) {
      int M = v + 8 * h;
      size_t idx = rowbase + (size_t)(qt * 16 + M) * DIM + hoff + t * 16 + ln;
      float val = ao[t][v] * sStat[32 + M];
      if (QVadd) val += bf2f(QVadd[idx]);
      O[idx] = f2bf(val);
    }
}

// ---------------- LayerNorm over [S, QD] per batch ----------------
__global__ void __launch_bounds__(256) k_layernorm(const unsigned short* __restrict__ X,
                                                   const float* __restrict__ w,
                                                   const float* __restrict__ bcoef,
                                                   unsigned short* __restrict__ Y) {
  __shared__ float s1[256];
  __shared__ float s2[256];
  const int tid = threadIdx.x;
  const int b = blockIdx.x;
  const size_t base = (size_t)b * SEQ * DIM;
  const int n = SEQ * DIM;
  float a = 0.0f, q = 0.0f;
  for (int i = tid; i < n; i += 256) { float v = bf2f(X[base + i]); a += v; q += v * v; }
  s1[tid] = a; s2[tid] = q;
  __syncthreads();
  for (int off = 128; off > 0; off >>= 1) {
    if (tid < off) { s1[tid] += s1[tid + off]; s2[tid] += s2[tid + off]; }
    __syncthreads();
  }
  const float invn = 1.0f / (float)n;
  float mean = s1[0] * invn;
  float var  = s2[0] * invn - mean * mean;
  float inv  = rsqrtf(var + 1e-5f);
  for (int i = tid; i < n; i += 256) {
    float v = (bf2f(X[base + i]) - mean) * inv * w[i] + bcoef[i];
    Y[base + i] = f2bf(v);
  }
}

// ---------------- host orchestration ----------------
static void gemm_launch(const unsigned short* A, const unsigned short* W, const float* bias,
                        const unsigned short* Res, unsigned short* Cb, float* Cf, int act,
                        hipStream_t s) {
  dim3 grid(DIM / 128, MROWS / 128);
  k_gemm<<<grid, dim3(128), 0, s>>>(A, W, bias, Res, Cb, Cf, MROWS, DIM, DIM, act);
}

extern "C" void kernel_launch(void* const* d_in, const int* in_sizes, int n_in,
                              void* d_out, int out_size, void* d_ws, size_t ws_size,
                              hipStream_t stream) {
  (void)in_sizes; (void)n_in; (void)out_size; (void)ws_size;

  const float* xin = (const float*)d_in[0];
  const float* yin = (const float*)d_in[1];
  const int w_idx[16] = {2, 4, 6, 8, 10, 14, 16, 18, 20, 22, 24, 26, 28, 30, 32, 34};
  const float* b_caq  = (const float*)d_in[3];
  const float* b_caqv = (const float*)d_in[5];
  const float* b_cak  = (const float*)d_in[7];
  const float* b_cakv = (const float*)d_in[9];
  const float* b_cao  = (const float*)d_in[11];
  const float* ln_w   = (const float*)d_in[12];
  const float* ln_b   = (const float*)d_in[13];
  const float* b_ff1  = (const float*)d_in[15];
  const float* b_ff2  = (const float*)d_in[17];
  const float* b_ff3  = (const float*)d_in[19];
  const float* b_s1q  = (const float*)d_in[21];
  const float* b_s1k  = (const float*)d_in[23];
  const float* b_s1v  = (const float*)d_in[25];
  const float* b_s1o  = (const float*)d_in[27];
  const float* b_s2q  = (const float*)d_in[29];
  const float* b_s2k  = (const float*)d_in[31];
  const float* b_s2v  = (const float*)d_in[33];
  const float* b_s2o  = (const float*)d_in[35];

  char* ws = (char*)d_ws;
  unsigned short* wb[16];
  for (int i = 0; i < 16; ++i) wb[i] = (unsigned short*)(ws + (size_t)i * 2097152);
  unsigned short* act[8];
  for (int i = 0; i < 8; ++i)
    act[i] = (unsigned short*)(ws + 33554432 + (size_t)i * 16777216);

  const int nAct = MROWS * DIM;        // 8388608
  const int nW   = DIM * DIM;          // 1048576

  // converts
  k_f2b<<<dim3((nAct + 255) / 256), dim3(256), 0, stream>>>(xin, act[0], nAct);
  k_f2b<<<dim3((nAct + 255) / 256), dim3(256), 0, stream>>>(yin, act[1], nAct);
  for (int i = 0; i < 16; ++i)
    k_f2b<<<dim3((nW + 255) / 256), dim3(256), 0, stream>>>((const float*)d_in[w_idx[i]], wb[i], nW);

  const size_t attn_shmem = 65536 + 32768 + 16384 + 256;
  const dim3 attn_grid(BATCH * HEADS * (SEQ / 16));
  const float sa_scale = 0.08838834764831845f; // 1/sqrt(128)

  // ---- cross attention ----
  gemm_launch(act[0], wb[0], b_caq,  nullptr, act[2], nullptr, 0, stream); // q
  gemm_launch(act[0], wb[1], b_caqv, nullptr, act[3], nullptr, 0, stream); // qv
  gemm_launch(act[1], wb[2], b_cak,  nullptr, act[4], nullptr, 0, stream); // k
  gemm_launch(act[1], wb[3], b_cakv, nullptr, act[5], nullptr, 0, stream); // kv
  k_attn<<<attn_grid, dim3(32), attn_shmem, stream>>>(act[2], act[4], act[5], act[3], act[6], 1.0f);
  gemm_launch(act[6], wb[4], b_cao, act[0] /* +x */, act[2], nullptr, 0, stream); // h1

  // ---- layernorm over [S, QD] ----
  k_layernorm<<<dim3(BATCH), dim3(256), 0, stream>>>(act[2], ln_w, ln_b, act[3]); // hn

  // ---- feed-forward with leaky_relu, residual ----
  gemm_launch(act[3], wb[5], b_ff1, nullptr, act[4], nullptr, 1, stream);
  gemm_launch(act[4], wb[6], b_ff2, nullptr, act[5], nullptr, 1, stream);
  gemm_launch(act[5], wb[7], b_ff3, act[3] /* +hn */, act[6], nullptr, 1, stream); // h2

  // ---- self attention 1 ----
  gemm_launch(act[6], wb[8],  b_s1q, nullptr, act[0], nullptr, 0, stream);
  gemm_launch(act[6], wb[9],  b_s1k, nullptr, act[1], nullptr, 0, stream);
  gemm_launch(act[6], wb[10], b_s1v, nullptr, act[2], nullptr, 0, stream);
  k_attn<<<attn_grid, dim3(32), attn_shmem, stream>>>(act[0], act[1], act[2], nullptr, act[4], sa_scale);
  gemm_launch(act[4], wb[11], b_s1o, nullptr, act[5], nullptr, 0, stream); // h3

  // ---- self attention 2 ----
  gemm_launch(act[5], wb[12], b_s2q, nullptr, act[0], nullptr, 0, stream);
  gemm_launch(act[5], wb[13], b_s2k, nullptr, act[1], nullptr, 0, stream);
  gemm_launch(act[5], wb[14], b_s2v, nullptr, act[2], nullptr, 0, stream);
  k_attn<<<attn_grid, dim3(32), attn_shmem, stream>>>(act[0], act[1], act[2], nullptr, act[4], sa_scale);
  gemm_launch(act[4], wb[15], b_s2o, nullptr, nullptr, (float*)d_out, 0, stream); // fp32 out
}
